// MyGCN_batch_norm_5102421148074
// MI455X (gfx1250) — matrix operand order repaired and checked by reading, not compiled, not run on hardware
//
#include <hip/hip_runtime.h>
#include <stdint.h>

// ---------------------------------------------------------------------------
// GCN on MI455X (gfx1250): bf16 WMMA for feature + adjacency GEMMs,
// BN + bias + dropout fused into the adjacency-GEMM epilogue via LDS.
// Layer-0 GEMM streams x (435 MB f32) exactly once, converting to bf16 in
// registers; adj lives as an 11.8 MB bf16 matrix resident in L2.
// ---------------------------------------------------------------------------

#define N_NODES 2371
#define NPAD    2432           // 152*16 row tiles, 76*32 K steps (even # pairs)
#define BATCH   32
#define CPAD    16             // per-layer feature width padded to WMMA N=16
#define NCOLS   (BATCH * CPAD) // 512 columns in the adjacency GEMM
#define KMAIN0  1408           // 44*32: guaranteed-in-bounds K for layer 0
#define KPAD0   1440           // 1433 padded to a multiple of 32
#define BN_EPS  1e-5f

typedef __attribute__((ext_vector_type(16))) __bf16    v16bf;
typedef __attribute__((ext_vector_type(8)))  float     v8f;
typedef __attribute__((ext_vector_type(4)))  uint32_t  u32x4;

union BFTile {
  v16bf v;
  u32x4 q[2];
  uint32_t d[8];
  unsigned short u[16];
  __bf16 h[16];
};

__device__ __forceinline__ unsigned short f2bf(float f) {
  uint32_t u = __builtin_bit_cast(uint32_t, f);
  u += 0x7FFFu + ((u >> 16) & 1u);           // round-to-nearest-even
  return (unsigned short)(u >> 16);
}

__device__ __forceinline__ uint32_t hashu(uint32_t x) {
  x ^= x >> 16; x *= 0x7feb352du;
  x ^= x >> 15; x *= 0x846ca68bu;
  x ^= x >> 16;
  return x;
}

// 16B-aligned tile load: halves p[0..7] and p[16..23] (2x global_load_b128)
__device__ __forceinline__ void loadB16(const unsigned short* p, BFTile& B) {
  B.q[0] = *(const u32x4*)(p);
  B.q[1] = *(const u32x4*)(p + 16);
}

// ---------------------------------------------------------------------------
// adj (f32 [2371,2371]) -> adjbf (bf16 [NPAD,NPAD], zero padded)
// ---------------------------------------------------------------------------
__global__ void cvt_adj_kernel(const float* __restrict__ adj,
                               unsigned short* __restrict__ adjbf) {
  size_t i = (size_t)blockIdx.x * blockDim.x + threadIdx.x;
  if (i >= (size_t)NPAD * NPAD) return;
  int r = (int)(i / NPAD), c = (int)(i % NPAD);
  float v = (r < N_NODES && c < N_NODES) ? adj[(size_t)r * N_NODES + c] : 0.f;
  adjbf[i] = f2bf(v);
}

// W (f32 [Fin,Fout]) -> WT (bf16 [16][Kpad], transposed + zero padded)
__global__ void cvt_w_kernel(const float* __restrict__ W,
                             unsigned short* __restrict__ WT,
                             int Fin, int Fout, int Kpad) {
  int i = blockIdx.x * blockDim.x + threadIdx.x;
  if (i >= 16 * Kpad) return;
  int gc = i / Kpad, k = i % Kpad;
  float v = (gc < Fout && k < Fin) ? W[(size_t)k * Fout + gc] : 0.f;
  WT[i] = f2bf(v);
}

// ---------------------------------------------------------------------------
// Layer-0 feature GEMM: S[(n,b), g] = sum_k bf16(x[b,n,k]) * W[k,g]
// Unconditional f32 loads in the main loop (k < 1408 always in bounds),
// cndmask-based tail; software pipelined (even/odd); one wave per 16-row tile.
// Output transposed: S_T[c=(b*16+g)][n].
// ---------------------------------------------------------------------------
__device__ __forceinline__ void loadAx(const float* __restrict__ row,
                                       int kbase, int koff, BFTile& A) {
  #pragma unroll
  for (int j = 0; j < 8; ++j) {
    A.h[j]     = (__bf16)row[kbase + koff + j];
    A.h[8 + j] = (__bf16)row[kbase + 16 + koff + j];
  }
}

__global__ void feat_gemm_x_kernel(const float* __restrict__ x,
                                   const unsigned short* __restrict__ WT,
                                   unsigned short* __restrict__ S_T) {
  const int lane  = threadIdx.x & 31;
  const int wave  = threadIdx.x >> 5;
  const int tile  = blockIdx.x * 4 + wave;   // (NPAD*BATCH)/16 tiles
  const int m0    = tile * 16;
  const int mrow  = m0 + (lane & 15);
  const int koff  = (lane < 16) ? 0 : 8;     // A lane K-offset (ISA layout)
  const int kboff = (lane < 16) ? 0 : 8;     // B lane K-offset (ISA layout)
  const int g     = lane & 15;

  const int n = mrow >> 5, b = mrow & 31;
  const int n_ld = (n < N_NODES) ? n : 0;    // clamp: loads always legal
  const float* row = x + ((size_t)b * N_NODES + n_ld) * 1433;
  const unsigned short* wrow = WT + (size_t)g * KPAD0;

  v8f acc;
  #pragma unroll
  for (int r = 0; r < 8; ++r) acc[r] = 0.f;

  BFTile Ae, Be, Ao, Bo;
  loadAx(row, 0, koff, Ae);
  loadB16(wrow + kboff, Be);

  for (int k0 = 0; k0 < KMAIN0; k0 += 64) {          // 22 pipelined pairs
    loadAx(row, k0 + 32, koff, Ao);
    loadB16(wrow + k0 + 32 + kboff, Bo);
    acc = __builtin_amdgcn_wmma_f32_16x16x32_bf16(false, Ae.v, false, Be.v,
                                                  (short)0, acc, false, false);
    if (k0 + 64 < KMAIN0) {
      loadAx(row, k0 + 64, koff, Ae);
      loadB16(wrow + k0 + 64 + kboff, Be);
    }
    acc = __builtin_amdgcn_wmma_f32_16x16x32_bf16(false, Ao.v, false, Bo.v,
                                                  (short)0, acc, false, false);
  }

  // Tail: K = 1408..1439 (valid < 1433). Selects only -- never predicated loads.
  {
    BFTile At, Bt;
    #pragma unroll
    for (int j = 0; j < 8; ++j) {
      int ka = KMAIN0 + koff + j;            // 1408..1423: always valid
      int kb = KMAIN0 + 16 + koff + j;       // 1424..1439: maybe invalid
      At.h[j] = (__bf16)row[ka];
      float fv = row[(kb < 1433) ? kb : 1432];
      At.h[8 + j] = (kb < 1433) ? (__bf16)fv : (__bf16)0.f;
    }
    loadB16(wrow + KMAIN0 + kboff, Bt);
    acc = __builtin_amdgcn_wmma_f32_16x16x32_bf16(false, At.v, false, Bt.v,
                                                  (short)0, acc, false, false);
  }

  #pragma unroll
  for (int r = 0; r < 8; ++r) {
    int M = r + ((lane >> 4) << 3);
    int m = m0 + M;
    int nn = m >> 5, bb = m & 31;
    unsigned short val = (nn < N_NODES) ? f2bf(acc[r]) : (unsigned short)0;
    S_T[(size_t)(bb * CPAD + g) * NPAD + nn] = val;
  }
}

// ---------------------------------------------------------------------------
// Layers >=1 feature GEMM: K = 16 (zero-padded to 32) -> one WMMA per tile.
// ---------------------------------------------------------------------------
__global__ void feat_gemm_act_kernel(const unsigned short* __restrict__ act,
                                     const unsigned short* __restrict__ WT,
                                     unsigned short* __restrict__ S_T) {
  const int lane  = threadIdx.x & 31;
  const int wave  = threadIdx.x >> 5;
  const int tile  = blockIdx.x * 4 + wave;
  const int m0    = tile * 16;
  const int mrow  = m0 + (lane & 15);
  const int koff  = (lane < 16) ? 0 : 8;
  const int kboff = (lane < 16) ? 0 : 8;
  const int g     = lane & 15;

  BFTile A, B;
  A.q[0] = *(const u32x4*)(act + (size_t)mrow * CPAD + koff);  // K 0..15 slice
  #pragma unroll
  for (int j = 4; j < 8; ++j) A.d[j] = 0u;                     // K 16..31 = 0
  loadB16(WT + (size_t)g * 32 + kboff, B);

  v8f acc;
  #pragma unroll
  for (int r = 0; r < 8; ++r) acc[r] = 0.f;
  acc = __builtin_amdgcn_wmma_f32_16x16x32_bf16(false, A.v, false, B.v,
                                                (short)0, acc, false, false);

  #pragma unroll
  for (int r = 0; r < 8; ++r) {
    int M = r + ((lane >> 4) << 3);
    int m = m0 + M;
    int nn = m >> 5, bb = m & 31;
    S_T[(size_t)(bb * CPAD + g) * NPAD + nn] = f2bf(acc[r]);   // padded rows are 0 already
  }
}

// ---------------------------------------------------------------------------
// Adjacency GEMM + bias + (BN) + (dropout), fused; software-pipelined K loop.
// One block = one 16-node row block x ALL 512 columns (8 waves x 4 tiles);
// BN stats per node via LDS ds_add_f32.
// ---------------------------------------------------------------------------
__device__ __forceinline__ void adj_load(const unsigned short* __restrict__ arow,
                                         const unsigned short* __restrict__ b0,
                                         const unsigned short* __restrict__ b1,
                                         const unsigned short* __restrict__ b2,
                                         const unsigned short* __restrict__ b3,
                                         int k0, int koff, int kboff,
                                         BFTile& A, BFTile& B0, BFTile& B1,
                                         BFTile& B2, BFTile& B3) {
  A.q[0] = *(const u32x4*)(arow + k0 + koff);
  A.q[1] = *(const u32x4*)(arow + k0 + 16 + koff);
  loadB16(b0 + k0 + kboff, B0);
  loadB16(b1 + k0 + kboff, B1);
  loadB16(b2 + k0 + kboff, B2);
  loadB16(b3 + k0 + kboff, B3);
}

__device__ __forceinline__ void wmma4(const BFTile& A, const BFTile& B0,
                                      const BFTile& B1, const BFTile& B2,
                                      const BFTile& B3, v8f (&acc)[4]) {
  acc[0] = __builtin_amdgcn_wmma_f32_16x16x32_bf16(false, A.v, false, B0.v,
                                                   (short)0, acc[0], false, false);
  acc[1] = __builtin_amdgcn_wmma_f32_16x16x32_bf16(false, A.v, false, B1.v,
                                                   (short)0, acc[1], false, false);
  acc[2] = __builtin_amdgcn_wmma_f32_16x16x32_bf16(false, A.v, false, B2.v,
                                                   (short)0, acc[2], false, false);
  acc[3] = __builtin_amdgcn_wmma_f32_16x16x32_bf16(false, A.v, false, B3.v,
                                                   (short)0, acc[3], false, false);
}

__global__ void adj_gemm_fused_kernel(const unsigned short* __restrict__ adjbf,
                                      const unsigned short* __restrict__ S_T,
                                      const float* __restrict__ bias,
                                      const float* __restrict__ gamma,  // null => no BN
                                      const float* __restrict__ beta,
                                      unsigned short* __restrict__ act_out,
                                      float* __restrict__ out_f32,
                                      int Fout, int layer, int do_dropout) {
  __shared__ float s_sum[16];
  __shared__ float s_sq[16];

  const int lane  = threadIdx.x & 31;
  const int wave  = threadIdx.x >> 5;
  const int n0    = blockIdx.x * 16;
  const int g     = lane & 15;
  const int koff  = (lane < 16) ? 0 : 8;
  const int kboff = (lane < 16) ? 0 : 8;

  v8f acc[4];
  #pragma unroll
  for (int t = 0; t < 4; ++t)
    #pragma unroll
    for (int r = 0; r < 8; ++r) acc[t][r] = 0.f;

  const unsigned short* arow = adjbf + (size_t)(n0 + g) * NPAD;
  const unsigned short* b0 = S_T + (size_t)((wave * 4 + 0) * CPAD + g) * NPAD;
  const unsigned short* b1 = S_T + (size_t)((wave * 4 + 1) * CPAD + g) * NPAD;
  const unsigned short* b2 = S_T + (size_t)((wave * 4 + 2) * CPAD + g) * NPAD;
  const unsigned short* b3 = S_T + (size_t)((wave * 4 + 3) * CPAD + g) * NPAD;

  BFTile Ae, Be0, Be1, Be2, Be3;
  BFTile Ao, Bo0, Bo1, Bo2, Bo3;
  adj_load(arow, b0, b1, b2, b3, 0, koff, kboff, Ae, Be0, Be1, Be2, Be3);

  for (int k0 = 0; k0 < NPAD; k0 += 64) {            // 38 pipelined pairs
    __builtin_prefetch(arow + k0 + 128, 0, 1);
    adj_load(arow, b0, b1, b2, b3, k0 + 32, koff, kboff, Ao, Bo0, Bo1, Bo2, Bo3);
    wmma4(Ae, Be0, Be1, Be2, Be3, acc);
    if (k0 + 64 < NPAD)
      adj_load(arow, b0, b1, b2, b3, k0 + 64, koff, kboff, Ae, Be0, Be1, Be2, Be3);
    wmma4(Ao, Bo0, Bo1, Bo2, Bo3, acc);
  }

  const bool  gvalid = (g < Fout);
  const float bv     = gvalid ? bias[g] : 0.f;

  if (gamma) {  // BN layers: per-node stats over (b, g<Fout) via LDS
    if (threadIdx.x < 16) { s_sum[threadIdx.x] = 0.f; s_sq[threadIdx.x] = 0.f; }
    __syncthreads();
    if (gvalid) {
      #pragma unroll
      for (int r = 0; r < 8; ++r) {
        const int M = r + ((lane >> 4) << 3);
        if (n0 + M < N_NODES) {
          float ps = 0.f, pq = 0.f;
          #pragma unroll
          for (int t = 0; t < 4; ++t) {
            float v = acc[t][r] + bv;
            ps += v; pq += v * v;
          }
          atomicAdd(&s_sum[M], ps);
          atomicAdd(&s_sq[M], pq);
        }
      }
    }
    __syncthreads();
  }

  const float inv_count = 1.f / (float)(BATCH * Fout);
  #pragma unroll
  for (int t = 0; t < 4; ++t) {
    const int bg = wave * 4 + t;
    #pragma unroll
    for (int r = 0; r < 8; ++r) {
      const int  M     = r + ((lane >> 4) << 3);
      const int  n     = n0 + M;
      const bool valid = gvalid && (n < N_NODES);
      float v = acc[t][r] + bv;
      if (gamma && valid) {
        float mean = s_sum[M] * inv_count;
        float var  = s_sq[M] * inv_count - mean * mean;
        v = (v - mean) * rsqrtf(var + BN_EPS) * gamma[n] + beta[n];
      }
      if (do_dropout) {
        uint32_t h = hashu(((uint32_t)layer << 24) ^
                           ((uint32_t)n * 512u + (uint32_t)(bg * CPAD + g)));
        v = (h & 0x10000u) ? v * 2.0f : 0.f;   // p = 0.5, scale 1/keep = 2
      }
      if (out_f32) {
        if (valid) out_f32[((size_t)bg * N_NODES + n) * Fout + g] = v;
      } else {
        act_out[((size_t)n * BATCH + bg) * CPAD + g] =
            valid ? f2bf(v) : (unsigned short)0;  // keep padding exactly zero
      }
    }
  }
}

// ---------------------------------------------------------------------------
// Host orchestration
// ---------------------------------------------------------------------------
extern "C" void kernel_launch(void* const* d_in, const int* in_sizes, int n_in,
                              void* d_out, int out_size, void* d_ws, size_t ws_size,
                              hipStream_t stream) {
  static const int DIMS[11] = {1433, 12, 11, 10, 9, 8, 7, 6, 5, 4, 7};

  const float* x   = (const float*)d_in[0];
  const float* adj = (const float*)d_in[1];
  const float* W[10];  const float* bs[10];
  const float* gam[7]; const float* bet[7];
  for (int i = 0; i < 10; ++i) { W[i]  = (const float*)d_in[2 + i];
                                 bs[i] = (const float*)d_in[12 + i]; }
  for (int i = 0; i < 7;  ++i) { gam[i] = (const float*)d_in[22 + i];
                                 bet[i] = (const float*)d_in[29 + i]; }

  char* ws = (char*)d_ws;
  size_t off = 0;
  auto walloc = [&](size_t bytes) -> void* {
    void* p = ws + off;
    off = (off + bytes + 255) & ~(size_t)255;
    return p;
  };

  unsigned short* adjbf = (unsigned short*)walloc((size_t)NPAD * NPAD * 2);   // 11.8 MB
  unsigned short* S_T   = (unsigned short*)walloc((size_t)NCOLS * NPAD * 2);  //  2.5 MB
  unsigned short* actA  = (unsigned short*)walloc((size_t)NPAD * BATCH * CPAD * 2);
  unsigned short* actB  = (unsigned short*)walloc((size_t)NPAD * BATCH * CPAD * 2);
  unsigned short* WT[10];
  for (int i = 0; i < 10; ++i) {
    int kp = (i == 0) ? KPAD0 : 32;
    WT[i] = (unsigned short*)walloc((size_t)16 * kp * 2);
  }

  {
    size_t total = (size_t)NPAD * NPAD;
    cvt_adj_kernel<<<(int)((total + 255) / 256), 256, 0, stream>>>(adj, adjbf);
  }
  for (int i = 0; i < 10; ++i) {
    int kp = (i == 0) ? KPAD0 : 32;
    int total = 16 * kp;
    cvt_w_kernel<<<(total + 255) / 256, 256, 0, stream>>>(W[i], WT[i],
                                                          DIMS[i], DIMS[i + 1], kp);
  }

  const int featBlocks = (NPAD * BATCH / 16) / 4;   // 4 waves per block
  const int adjBlocks  = NPAD / 16;

  unsigned short* cur = actA;
  unsigned short* nxt = actB;
  for (int l = 0; l < 10; ++l) {
    if (l == 0) {
      feat_gemm_x_kernel<<<featBlocks, 128, 0, stream>>>(x, WT[0], S_T);
    } else {
      feat_gemm_act_kernel<<<featBlocks, 128, 0, stream>>>(cur, WT[l], S_T);
    }

    const float* g  = (l < 7) ? gam[l] : nullptr;
    const float* bt = (l < 7) ? bet[l] : nullptr;
    unsigned short* ao = (l == 9) ? nullptr : nxt;
    float*          of = (l == 9) ? (float*)d_out : nullptr;
    adj_gemm_fused_kernel<<<adjBlocks, 256, 0, stream>>>(
        adjbf, S_T, bs[l], g, bt, ao, of, DIMS[l + 1], l, (l < 9) ? 1 : 0);

    unsigned short* tmp = cur; cur = nxt; nxt = tmp;
  }
}
